// Block_2010044694563
// MI455X (gfx1250) — matrix operand order repaired and checked by reading, not compiled, not run on hardware
//
#include <hip/hip_runtime.h>
#include <math.h>

typedef __bf16 bf16;
typedef bf16 v16bf __attribute__((ext_vector_type(16)));
typedef bf16 v8bf  __attribute__((ext_vector_type(8)));
typedef float v8f  __attribute__((ext_vector_type(8)));

union BF16Frag { v16bf v; v8bf h[2]; };

#define BATCH 4
#define SEQ   2048
#define DIM   768
#define NH    12
#define HS    64
#define NT    (BATCH * SEQ)   // 8192 tokens

__device__ __forceinline__ float gelu_exact(float v) {
    return 0.5f * v * (1.0f + erff(v * 0.70710678118654752f));
}

// ---------------------------------------------------------------------------
// LayerNorm (fp32 in) -> bf16 out, one block per row of 768
// ---------------------------------------------------------------------------
__global__ __launch_bounds__(256) void ln_to_bf16(
    const float* __restrict__ x, const float* __restrict__ g,
    const float* __restrict__ b, bf16* __restrict__ out)
{
    const int row = blockIdx.x;
    const int tid = threadIdx.x;
    const float* xr = x + (size_t)row * DIM;
    __shared__ float red[256];

    float s = 0.f;
    for (int i = tid; i < DIM; i += 256) s += xr[i];
    red[tid] = s; __syncthreads();
    for (int st = 128; st > 0; st >>= 1) { if (tid < st) red[tid] += red[tid + st]; __syncthreads(); }
    const float mu = red[0] * (1.0f / DIM);
    __syncthreads();

    float v = 0.f;
    for (int i = tid; i < DIM; i += 256) { float d = xr[i] - mu; v += d * d; }
    red[tid] = v; __syncthreads();
    for (int st = 128; st > 0; st >>= 1) { if (tid < st) red[tid] += red[tid + st]; __syncthreads(); }
    const float rstd = rsqrtf(red[0] * (1.0f / DIM) + 1e-5f);

    bf16* orow = out + (size_t)row * DIM;
    for (int i = tid; i < DIM; i += 256)
        orow[i] = (bf16)((xr[i] - mu) * rstd * g[i] + b[i]);
}

// ---------------------------------------------------------------------------
// GEMM: C[MxN] = A[MxK](bf16) @ W[KxN](f32->bf16) with fused epilogue.
// Block tile 64x128, K-tile 32; 8 waves (2x4), each wave owns a 32x32 output
// tile: 4 WMMA accumulators fed by 2 A-frags + 2 B-frags per K-tile.
// ---------------------------------------------------------------------------
__global__ __launch_bounds__(256) void gemm_bf16_wmma(
    const bf16*  __restrict__ A,
    const float* __restrict__ W,
    const float* __restrict__ bias,   // [N] or null
    const float* __restrict__ resid,  // [MxN] f32 or null
    float*       __restrict__ outF,   // [MxN] f32 or null
    bf16*        __restrict__ outB,   // [MxN] bf16 or null
    int M, int N, int K, int gelu)
{
    __shared__ bf16 As[64][40];    // [m][k], +8 pad
    __shared__ bf16 Bs[128][40];   // [n][k] (W transposed), +8 pad

    const int tid  = threadIdx.x;
    const int lane = tid & 31, w = tid >> 5;
    const int half = lane >> 4, lo = lane & 15;
    const int wm = w >> 2, wn = w & 3;            // 2x4 wave grid
    const int m0 = blockIdx.y * 64, n0 = blockIdx.x * 128;

    v8f acc00 = {}, acc01 = {}, acc10 = {}, acc11 = {};

    // staging maps (hoisted base pointers; bump per K-tile, no in-loop muls)
    const int arow = tid >> 2, akk = (tid & 3) * 8;       // A: 64x32, 8 bf16/thread
    const int bn   = tid & 127, bkb = (tid >> 7) * 16;    // B: 128x32, 16 f32/thread
    const bf16*  aptr = A + (size_t)(m0 + arow) * K + akk;
    const float* wptr = W + (size_t)bkb * N + n0 + bn;
    const size_t wstep = (size_t)32 * N;

    for (int kt = 0; kt < K; kt += 32) {
        __syncthreads();
        // stage A: vectorized 16B load
        *(v8bf*)&As[arow][akk] = *(const v8bf*)aptr;
        // stage W transposed: Bs[n][k] = (bf16)W[kt+k][n0+n]; coalesced across n
        #pragma unroll
        for (int i = 0; i < 16; ++i)
            Bs[bn][bkb + i] = (bf16)wptr[(size_t)i * N];
        // prefetch next K-tile of W into cache hierarchy
        if (kt + 32 < K)
            __builtin_prefetch(wptr + wstep, 0, 3);
        aptr += 32;
        wptr += wstep;
        __syncthreads();

        BF16Frag b0, b1;
        b0.h[0] = *(const v8bf*)&Bs[wn * 32 + lo][half * 8];
        b0.h[1] = *(const v8bf*)&Bs[wn * 32 + lo][16 + half * 8];
        b1.h[0] = *(const v8bf*)&Bs[wn * 32 + 16 + lo][half * 8];
        b1.h[1] = *(const v8bf*)&Bs[wn * 32 + 16 + lo][16 + half * 8];

        BF16Frag a0, a1;
        a0.h[0] = *(const v8bf*)&As[wm * 32 + lo][half * 8];
        a0.h[1] = *(const v8bf*)&As[wm * 32 + lo][16 + half * 8];
        a1.h[0] = *(const v8bf*)&As[wm * 32 + 16 + lo][half * 8];
        a1.h[1] = *(const v8bf*)&As[wm * 32 + 16 + lo][16 + half * 8];

        acc00 = __builtin_amdgcn_wmma_f32_16x16x32_bf16(false, a0.v, false, b0.v,
                                                        (short)0, acc00, false, false);
        acc01 = __builtin_amdgcn_wmma_f32_16x16x32_bf16(false, a0.v, false, b1.v,
                                                        (short)0, acc01, false, false);
        acc10 = __builtin_amdgcn_wmma_f32_16x16x32_bf16(false, a1.v, false, b0.v,
                                                        (short)0, acc10, false, false);
        acc11 = __builtin_amdgcn_wmma_f32_16x16x32_bf16(false, a1.v, false, b1.v,
                                                        (short)0, acc11, false, false);
    }

    #pragma unroll
    for (int sm = 0; sm < 2; ++sm) {
        #pragma unroll
        for (int sn = 0; sn < 2; ++sn) {
            v8f acc = sm ? (sn ? acc11 : acc10) : (sn ? acc01 : acc00);
            #pragma unroll
            for (int r = 0; r < 8; ++r) {
                const int gr = m0 + wm * 32 + sm * 16 + r + 8 * half;
                const int gc = n0 + wn * 32 + sn * 16 + lo;
                float v = acc[r];
                if (bias)  v += bias[gc];
                if (gelu)  v  = gelu_exact(v);
                const size_t off = (size_t)gr * N + gc;
                if (resid) v += resid[off];
                if (outF)  outF[off] = v;
                if (outB)  outB[off] = (bf16)v;
            }
        }
    }
}

// ---------------------------------------------------------------------------
// Flash-style attention: one block per (batch, head, 64-query tile).
// Scores and PV via v_wmma_f32_16x16x32_bf16; online softmax in fp32.
// ---------------------------------------------------------------------------
__global__ __launch_bounds__(256) void attention_wmma(
    const bf16* __restrict__ Q, const bf16* __restrict__ Kb,
    const bf16* __restrict__ Vb, const int* __restrict__ amask,
    bf16* __restrict__ Out)
{
    __shared__ bf16 Qs[64][72];    // [q][d]
    __shared__ bf16 Ks[64][72];    // [key][d]  (acts as B-tile for Q@K^T)
    __shared__ bf16 Vt[64][72];    // [d][key]  (acts as B-tile for P@V)
    __shared__ bf16 Pt[64][72];    // [q][key]  bf16 probabilities
    __shared__ float St[64][68];   // fp32 scores
    __shared__ float mrow[64], lrow[64], alph[64], mbias[64];

    const int tid  = threadIdx.x;
    const int lane = tid & 31, w = tid >> 5;
    const int half = lane >> 4, lo = lane & 15;
    const int wm = w >> 2, wn = w & 3;
    const int q0 = blockIdx.x * 64;
    const int h  = blockIdx.y;
    const int b  = blockIdx.z;
    const float scl = 0.03608439182435161f;   // 768^-0.5 (embed-dim scaling, per reference)

    {   // stage Q tile
        const int row = tid >> 2, ds = (tid & 3) * 16;
        const bf16* gp = Q + ((size_t)(b * SEQ + q0 + row) * DIM + h * HS + ds);
        *(v8bf*)&Qs[row][ds]     = *(const v8bf*)gp;
        *(v8bf*)&Qs[row][ds + 8] = *(const v8bf*)(gp + 8);
    }
    if (tid < 64) { mrow[tid] = -INFINITY; lrow[tid] = 0.f; }

    v8f o0 = {}; v8f o1 = {};

    for (int k0 = 0; k0 < SEQ; k0 += 64) {
        __syncthreads();
        {   // stage K chunk direct, V chunk transposed
            const int row = tid >> 2, ds = (tid & 3) * 16;
            const bf16* kp = Kb + ((size_t)(b * SEQ + k0 + row) * DIM + h * HS + ds);
            *(v8bf*)&Ks[row][ds]     = *(const v8bf*)kp;
            *(v8bf*)&Ks[row][ds + 8] = *(const v8bf*)(kp + 8);
            const bf16* vp = Vb + ((size_t)(b * SEQ + k0 + row) * DIM + h * HS + ds);
            #pragma unroll
            for (int i = 0; i < 16; ++i) Vt[ds + i][row] = vp[i];
        }
        if (tid < 64)
            mbias[tid] = (amask[b * SEQ + k0 + tid] == 0) ? -INFINITY : 0.f;
        __syncthreads();

        // ---- scores S = Q @ K^T (per-wave 32x16 strip) ----
        v8f s0 = {}; v8f s1 = {};
        #pragma unroll
        for (int kk = 0; kk < HS; kk += 32) {
            BF16Frag bfr;
            bfr.h[0] = *(const v8bf*)&Ks[wn * 16 + lo][kk + half * 8];
            bfr.h[1] = *(const v8bf*)&Ks[wn * 16 + lo][kk + 16 + half * 8];
            BF16Frag a0f, a1f;
            a0f.h[0] = *(const v8bf*)&Qs[wm * 32 + lo][kk + half * 8];
            a0f.h[1] = *(const v8bf*)&Qs[wm * 32 + lo][kk + 16 + half * 8];
            a1f.h[0] = *(const v8bf*)&Qs[wm * 32 + 16 + lo][kk + half * 8];
            a1f.h[1] = *(const v8bf*)&Qs[wm * 32 + 16 + lo][kk + 16 + half * 8];
            s0 = __builtin_amdgcn_wmma_f32_16x16x32_bf16(false, a0f.v, false, bfr.v,
                                                         (short)0, s0, false, false);
            s1 = __builtin_amdgcn_wmma_f32_16x16x32_bf16(false, a1f.v, false, bfr.v,
                                                         (short)0, s1, false, false);
        }
        #pragma unroll
        for (int r = 0; r < 8; ++r) {
            St[wm * 32 + r + 8 * half][wn * 16 + lo]      = s0[r] * scl;
            St[wm * 32 + 16 + r + 8 * half][wn * 16 + lo] = s1[r] * scl;
        }
        __syncthreads();

        // ---- online softmax: one thread per query row ----
        if (tid < 64) {
            const int row = tid;
            const float mo = mrow[row];
            float rm = -INFINITY;
            #pragma unroll 8
            for (int c = 0; c < 64; ++c)
                rm = fmaxf(rm, St[row][c] + mbias[c]);
            const float mn = fmaxf(mo, rm);
            const float al = (mo == -INFINITY) ? 0.f : expf(mo - mn);
            float sum = 0.f;
            #pragma unroll 8
            for (int c = 0; c < 64; ++c) {
                const float s = St[row][c] + mbias[c];
                const float p = (mn == -INFINITY) ? 0.f : expf(s - mn);
                Pt[row][c] = (bf16)p;
                sum += p;
            }
            mrow[row] = mn;
            lrow[row] = lrow[row] * al + sum;
            alph[row] = al;
        }
        __syncthreads();

        // ---- rescale O fragments, then O += P @ V ----
        #pragma unroll
        for (int r = 0; r < 8; ++r) {
            o0[r] *= alph[wm * 32 + r + 8 * half];
            o1[r] *= alph[wm * 32 + 16 + r + 8 * half];
        }
        #pragma unroll
        for (int kk = 0; kk < 64; kk += 32) {
            BF16Frag bfr;
            bfr.h[0] = *(const v8bf*)&Vt[wn * 16 + lo][kk + half * 8];
            bfr.h[1] = *(const v8bf*)&Vt[wn * 16 + lo][kk + 16 + half * 8];
            BF16Frag a0f, a1f;
            a0f.h[0] = *(const v8bf*)&Pt[wm * 32 + lo][kk + half * 8];
            a0f.h[1] = *(const v8bf*)&Pt[wm * 32 + lo][kk + 16 + half * 8];
            a1f.h[0] = *(const v8bf*)&Pt[wm * 32 + 16 + lo][kk + half * 8];
            a1f.h[1] = *(const v8bf*)&Pt[wm * 32 + 16 + lo][kk + 16 + half * 8];
            o0 = __builtin_amdgcn_wmma_f32_16x16x32_bf16(false, a0f.v, false, bfr.v,
                                                         (short)0, o0, false, false);
            o1 = __builtin_amdgcn_wmma_f32_16x16x32_bf16(false, a1f.v, false, bfr.v,
                                                         (short)0, o1, false, false);
        }
    }

    // ---- normalize by l and write attn output bf16 (head-sliced layout) ----
    #pragma unroll
    for (int r = 0; r < 8; ++r) {
        {
            const int row = wm * 32 + r + 8 * half;
            const float li = lrow[row];
            const float vo = (li > 0.f) ? o0[r] / li : 0.f;
            Out[(size_t)(b * SEQ + q0 + row) * DIM + h * HS + wn * 16 + lo] = (bf16)vo;
        }
        {
            const int row = wm * 32 + 16 + r + 8 * half;
            const float li = lrow[row];
            const float vo = (li > 0.f) ? o1[r] / li : 0.f;
            Out[(size_t)(b * SEQ + q0 + row) * DIM + h * HS + wn * 16 + lo] = (bf16)vo;
        }
    }
}

// ---------------------------------------------------------------------------
extern "C" void kernel_launch(void* const* d_in, const int* in_sizes, int n_in,
                              void* d_out, int out_size, void* d_ws, size_t ws_size,
                              hipStream_t stream)
{
    (void)in_sizes; (void)n_in; (void)out_size; (void)ws_size;
    const float* x   = (const float*)d_in[0];
    const int*   am  = (const int*)d_in[1];
    const float* Wq  = (const float*)d_in[2];
    const float* Wk  = (const float*)d_in[3];
    const float* Wv  = (const float*)d_in[4];
    const float* Wo  = (const float*)d_in[5];
    const float* bo  = (const float*)d_in[6];
    const float* g1  = (const float*)d_in[7];
    const float* be1 = (const float*)d_in[8];
    const float* g2  = (const float*)d_in[9];
    const float* be2 = (const float*)d_in[10];
    const float* W1  = (const float*)d_in[11];
    const float* bf1 = (const float*)d_in[12];
    const float* W2  = (const float*)d_in[13];
    const float* bf2 = (const float*)d_in[14];
    float* out = (float*)d_out;

    const size_t nelem = (size_t)NT * DIM;
    char* ws = (char*)d_ws;
    bf16* hbf = (bf16*)ws;  ws += nelem * 2;   // LN1(x)
    bf16* qb  = (bf16*)ws;  ws += nelem * 2;   // Q  (later reused as LN2 out)
    bf16* kb  = (bf16*)ws;  ws += nelem * 2;   // K  (later reused as FFN hidden)
    bf16* vb  = (bf16*)ws;  ws += nelem * 2;   // V
    bf16* atn = (bf16*)ws;  ws += nelem * 2;   // attention output
    float* x1 = (float*)ws; ws += nelem * 4;   // post-attention residual

    dim3 blk(256);
    dim3 ggrid(DIM / 128, NT / 64);
    dim3 agrid(SEQ / 64, NH, BATCH);

    ln_to_bf16<<<NT, blk, 0, stream>>>(x, g1, be1, hbf);
    gemm_bf16_wmma<<<ggrid, blk, 0, stream>>>(hbf, Wq, nullptr, nullptr, nullptr, qb, NT, DIM, DIM, 0);
    gemm_bf16_wmma<<<ggrid, blk, 0, stream>>>(hbf, Wk, nullptr, nullptr, nullptr, kb, NT, DIM, DIM, 0);
    gemm_bf16_wmma<<<ggrid, blk, 0, stream>>>(hbf, Wv, nullptr, nullptr, nullptr, vb, NT, DIM, DIM, 0);
    attention_wmma<<<agrid, blk, 0, stream>>>(qb, kb, vb, am, atn);
    gemm_bf16_wmma<<<ggrid, blk, 0, stream>>>(atn, Wo, bo, x, x1, nullptr, NT, DIM, DIM, 0);
    // reuse qb as LN2 output, kb as FFN hidden (Q/K no longer needed)
    ln_to_bf16<<<NT, blk, 0, stream>>>(x1, g2, be2, qb);
    gemm_bf16_wmma<<<ggrid, blk, 0, stream>>>(qb, W1, bf1, nullptr, nullptr, kb, NT, DIM, DIM, 1);
    gemm_bf16_wmma<<<ggrid, blk, 0, stream>>>(kb, W2, bf2, x1, out, nullptr, NT, DIM, DIM, 0);
}
